// DNADecoder_180388627219
// MI455X (gfx1250) — matrix operand order repaired and checked by reading, not compiled, not run on hardware
//
#include <hip/hip_runtime.h>

// dims from the reference
constexpr int B_ = 64, T_ = 128, P_ = 1024, D_ = 512, F_ = 2048, V_ = 4096, L_ = 4;

typedef __attribute__((ext_vector_type(16))) __bf16 v16bf;
typedef __attribute__((ext_vector_type(8)))  __bf16 v8bf;
typedef __attribute__((ext_vector_type(8)))  float  v8f;

// ---------------- fp32 [K,N] -> bf16 [N,K] tiled transpose (LDS) ----------------
__global__ void k_transpose_bf16(const float* __restrict__ W, __bf16* __restrict__ WT,
                                 int K, int N) {
  __shared__ float tile[32][33];
  int kb = blockIdx.x * 32, nb = blockIdx.y * 32;
  int tx = threadIdx.x, ty = threadIdx.y;   // 32 x 8
  #pragma unroll
  for (int i = ty; i < 32; i += 8)
    tile[i][tx] = W[(size_t)(kb + i) * N + (nb + tx)];
  __syncthreads();
  #pragma unroll
  for (int i = ty; i < 32; i += 8)
    WT[(size_t)(nb + i) * K + (kb + tx)] = (__bf16)tile[tx][i];
}

// ---------------- flat fp32 -> bf16 ----------------
__global__ void k_f2b(const float* __restrict__ in, __bf16* __restrict__ out, int n) {
  int i = blockIdx.x * 256 + threadIdx.x;
  if (i < n) out[i] = (__bf16)in[i];
}

// ---------------- WMMA GEMM: C[64,N] = A[64,K](bf16) @ W[K,N], W stored as WT[N,K] bf16
// Each wave: 16 rows x 64 cols (4 n-tiles, A fragment reused). M fixed = 64.
// Grid: N/64 blocks of 128 threads (4 full waves/block) -> no partial waves (EXEC all-1).
// K loop is software-pipelined: chunk k+32 fragments load while chunk k WMMAs execute,
// so v_wmma never waits on a just-issued global_load.
__device__ __forceinline__ v16bf ldfragA(const __bf16* __restrict__ Arow, int k0, int s) {
  // A 16x32 bf16 layout: element e -> K = 16*(e/8) + 8*s + (e%8)
  v8bf a0 = *(const v8bf*)(Arow + k0 + 8 * s);
  v8bf a1 = *(const v8bf*)(Arow + k0 + 16 + 8 * s);
  return __builtin_shufflevector(a0, a1, 0, 1, 2, 3, 4, 5, 6, 7,
                                 8, 9, 10, 11, 12, 13, 14, 15);
}
__device__ __forceinline__ v16bf ldfragB(const __bf16* __restrict__ Brow, int k0, int s) {
  // B 32x16 bf16 layout: element e -> K = 16*(e/8) + 8*s + (e%8) (column n held per lane)
  v8bf b0 = *(const v8bf*)(Brow + k0 + 8 * s);
  v8bf b1 = *(const v8bf*)(Brow + k0 + 16 + 8 * s);
  return __builtin_shufflevector(b0, b1, 0, 1, 2, 3, 4, 5, 6, 7,
                                 8, 9, 10, 11, 12, 13, 14, 15);
}

__global__ void k_gemm_bf16(const __bf16* __restrict__ A, const __bf16* __restrict__ BT,
                            const float* __restrict__ bias,
                            float* __restrict__ Cf, __bf16* __restrict__ Cb,
                            int N, int K, int relu) {
  int wave = (blockIdx.x * blockDim.x + threadIdx.x) >> 5;
  int lane = threadIdx.x & 31;
  int mt = wave & 3;            // 64 rows -> 4 m-tiles
  int ng = wave >> 2;           // 64-col group
  int m0 = mt * 16, n0 = ng * 64;
  int lr = lane & 15, s = lane >> 4;

  v8f acc[4] = {};
  const __bf16* Arow = A + (size_t)(m0 + lr) * K;
  const __bf16* Bb[4];
  #pragma unroll
  for (int nt = 0; nt < 4; ++nt) Bb[nt] = BT + (size_t)(n0 + nt * 16 + lr) * K;

  // prologue: fragments for chunk 0
  v16bf a_cur = ldfragA(Arow, 0, s);
  v16bf b_cur[4];
  #pragma unroll
  for (int nt = 0; nt < 4; ++nt) b_cur[nt] = ldfragB(Bb[nt], 0, s);

  for (int k0 = 0; k0 < K - 32; k0 += 32) {
    // issue next chunk's loads first ...
    v16bf a_nxt = ldfragA(Arow, k0 + 32, s);
    v16bf b_nxt[4];
    #pragma unroll
    for (int nt = 0; nt < 4; ++nt) b_nxt[nt] = ldfragB(Bb[nt], k0 + 32, s);
    // ... then compute on the already-resident chunk
    #pragma unroll
    for (int nt = 0; nt < 4; ++nt)
      acc[nt] = __builtin_amdgcn_wmma_f32_16x16x32_bf16(
          false, a_cur, false, b_cur[nt], (short)0, acc[nt], false, false);
    a_cur = a_nxt;
    #pragma unroll
    for (int nt = 0; nt < 4; ++nt) b_cur[nt] = b_nxt[nt];
  }
  // epilogue chunk
  #pragma unroll
  for (int nt = 0; nt < 4; ++nt)
    acc[nt] = __builtin_amdgcn_wmma_f32_16x16x32_bf16(
        false, a_cur, false, b_cur[nt], (short)0, acc[nt], false, false);

  // D layout: VGPR r -> row = m0 + r + 8*s, col = n0 + nt*16 + lr
  #pragma unroll
  for (int nt = 0; nt < 4; ++nt) {
    int col = n0 + nt * 16 + lr;
    float bb = bias ? bias[col] : 0.0f;
    #pragma unroll
    for (int r = 0; r < 8; ++r) {
      int row = m0 + r + 8 * s;
      float v = acc[nt][r] + bb;
      if (relu) v = fmaxf(v, 0.0f);
      size_t idx = (size_t)row * N + col;
      if (Cf) Cf[idx] = v;
      if (Cb) Cb[idx] = (__bf16)v;
    }
  }
}

// ---------------- embed + positional encoding ----------------
__global__ void k_embed(const float* __restrict__ emb, const float* __restrict__ pe,
                        const int* __restrict__ tok, int t,
                        float* __restrict__ x, __bf16* __restrict__ xb) {
  int b = blockIdx.x, d = threadIdx.x;  // 512 threads
  float v = emb[(size_t)tok[b] * D_ + d] + pe[(size_t)t * D_ + d];
  x[(size_t)b * D_ + d] = v;
  xb[(size_t)b * D_ + d] = (__bf16)v;
}

// ---------------- residual add + LayerNorm (per row), writes fp32 + bf16 ----------------
__global__ void k_add_ln(const float* __restrict__ xin, const float* __restrict__ delta,
                         const float* __restrict__ g, const float* __restrict__ be,
                         float* __restrict__ xout, __bf16* __restrict__ xb) {
  int b = blockIdx.x, tid = threadIdx.x;  // 256 threads, D=512 -> 2 elems/thread
  __shared__ float red[256];
  __shared__ float stat[2];
  size_t i0 = (size_t)b * D_ + tid, i1 = i0 + 256;
  float v0 = xin[i0] + delta[i0];
  float v1 = xin[i1] + delta[i1];
  red[tid] = v0 + v1;
  __syncthreads();
  for (int s2 = 128; s2 > 0; s2 >>= 1) {
    if (tid < s2) red[tid] += red[tid + s2];
    __syncthreads();
  }
  if (tid == 0) stat[0] = red[0] * (1.0f / D_);
  __syncthreads();
  float mean = stat[0];
  float d0 = v0 - mean, d1 = v1 - mean;
  red[tid] = d0 * d0 + d1 * d1;
  __syncthreads();
  for (int s2 = 128; s2 > 0; s2 >>= 1) {
    if (tid < s2) red[tid] += red[tid + s2];
    __syncthreads();
  }
  if (tid == 0) stat[1] = red[0] * (1.0f / D_);
  __syncthreads();
  float inv = rsqrtf(stat[1] + 1e-5f);
  float o0 = d0 * inv * g[tid] + be[tid];
  float o1 = d1 * inv * g[tid + 256] + be[tid + 256];
  xout[i0] = o0; xout[i1] = o1;
  xb[i0] = (__bf16)o0; xb[i1] = (__bf16)o1;
}

// ---------------- softmax over V=4096 + argmax carry token ----------------
__global__ void k_softmax_argmax(const float* __restrict__ logits, float* __restrict__ out,
                                 int* __restrict__ tok, int t) {
  int b = blockIdx.x, tid = threadIdx.x;  // 256 threads, 16 elems/thread
  const float* row = logits + (size_t)b * V_;
  __shared__ float sv[256];
  __shared__ int   si[256];
  float xs[16];
  float mx = -3.4e38f; int mi = 0;
  #pragma unroll
  for (int j = 0; j < 16; ++j) {
    int v = tid + j * 256;
    xs[j] = row[v];
    if (xs[j] > mx) { mx = xs[j]; mi = v; }
  }
  sv[tid] = mx; si[tid] = mi;
  __syncthreads();
  for (int s2 = 128; s2 > 0; s2 >>= 1) {
    if (tid < s2) {
      float xo = sv[tid + s2]; int io = si[tid + s2];
      if (xo > sv[tid] || (xo == sv[tid] && io < si[tid])) { sv[tid] = xo; si[tid] = io; }
    }
    __syncthreads();
  }
  float gmax = sv[0];
  if (tid == 0) tok[b] = si[0];
  __syncthreads();
  float sum = 0.0f;
  #pragma unroll
  for (int j = 0; j < 16; ++j) {
    xs[j] = expf(xs[j] - gmax);   // keep exp in registers for the store pass
    sum += xs[j];
  }
  sv[tid] = sum;
  __syncthreads();
  for (int s2 = 128; s2 > 0; s2 >>= 1) {
    if (tid < s2) sv[tid] += sv[tid + s2];
    __syncthreads();
  }
  float inv = 1.0f / sv[0];
  float* orow = out + ((size_t)b * T_ + t) * V_;
  #pragma unroll
  for (int j = 0; j < 16; ++j) orow[tid + j * 256] = xs[j] * inv;
}

extern "C" void kernel_launch(void* const* d_in, const int* in_sizes, int n_in,
                              void* d_out, int out_size, void* d_ws, size_t ws_size,
                              hipStream_t stream) {
  (void)in_sizes; (void)n_in; (void)out_size; (void)ws_size;
  const float* protein  = (const float*)d_in[0];
  const int*   tok0     = (const int*)d_in[1];
  const float* proj1_w  = (const float*)d_in[2];
  const float* proj1_b  = (const float*)d_in[3];
  const float* proj2_w  = (const float*)d_in[4];
  const float* proj2_b  = (const float*)d_in[5];
  const float* emb      = (const float*)d_in[6];
  const float* sa_w     = (const float*)d_in[7];
  const float* sa_b     = (const float*)d_in[8];
  const float* ca_w     = (const float*)d_in[9];
  const float* ca_b     = (const float*)d_in[10];
  const float* ffn_w1   = (const float*)d_in[11];
  const float* ffn_b1   = (const float*)d_in[12];
  const float* ffn_w2   = (const float*)d_in[13];
  const float* ffn_b2   = (const float*)d_in[14];
  const float* ln_g     = (const float*)d_in[15];
  const float* ln_b     = (const float*)d_in[16];
  const float* out_w    = (const float*)d_in[17];
  const float* out_b    = (const float*)d_in[18];
  const float* pe       = (const float*)d_in[19];
  float* probs = (float*)d_out;

  // ---- workspace layout (≈34 MB) ----
  char* wsp = (char*)d_ws;
  size_t off = 0;
  auto alloc = [&](size_t bytes) -> char* {
    char* p = wsp + off; off += (bytes + 255) & ~(size_t)255; return p;
  };
  __bf16* saWvT  = (__bf16*)alloc((size_t)L_ * D_ * D_ * 2);
  __bf16* saWoT  = (__bf16*)alloc((size_t)L_ * D_ * D_ * 2);
  __bf16* caWvT  = (__bf16*)alloc((size_t)L_ * D_ * D_ * 2);
  __bf16* caWoT  = (__bf16*)alloc((size_t)L_ * D_ * D_ * 2);
  __bf16* fW1T   = (__bf16*)alloc((size_t)L_ * F_ * D_ * 2);
  __bf16* fW2T   = (__bf16*)alloc((size_t)L_ * D_ * F_ * 2);
  __bf16* outWT  = (__bf16*)alloc((size_t)V_ * D_ * 2);
  __bf16* p1T    = (__bf16*)alloc((size_t)(2 * D_) * P_ * 2);
  __bf16* p2T    = (__bf16*)alloc((size_t)D_ * (2 * D_) * 2);
  __bf16* protb  = (__bf16*)alloc((size_t)B_ * P_ * 2);
  __bf16* mem1b  = (__bf16*)alloc((size_t)B_ * D_ * 2);
  float*  caC    = (float*)alloc((size_t)L_ * B_ * D_ * 4);
  float*  x      = (float*)alloc((size_t)B_ * D_ * 4);
  __bf16* xb     = (__bf16*)alloc((size_t)B_ * D_ * 2);
  __bf16* t1b    = (__bf16*)alloc((size_t)B_ * F_ * 2);
  float*  t2     = (float*)alloc((size_t)B_ * D_ * 4);
  float*  logits = (float*)alloc((size_t)B_ * V_ * 4);
  int*    curtok = (int*)alloc((size_t)B_ * 4);

  dim3 tb(32, 8);
  auto TR = [&](const float* W, __bf16* WT, int K, int N) {
    k_transpose_bf16<<<dim3(K / 32, N / 32), tb, 0, stream>>>(W, WT, K, N);
  };
  auto GEMM = [&](const __bf16* A, const __bf16* BT, const float* bias,
                  float* Cf, __bf16* Cb, int N, int K, int relu) {
    k_gemm_bf16<<<dim3(N / 64), dim3(128), 0, stream>>>(A, BT, bias, Cf, Cb, N, K, relu);
  };

  // ---- one-time per call: weight transpose+quantize to bf16 [N,K] ----
  for (int l = 0; l < L_; ++l) {
    TR(sa_w + ((size_t)l * 4 + 2) * D_ * D_, saWvT + (size_t)l * D_ * D_, D_, D_);
    TR(sa_w + ((size_t)l * 4 + 3) * D_ * D_, saWoT + (size_t)l * D_ * D_, D_, D_);
    TR(ca_w + ((size_t)l * 4 + 2) * D_ * D_, caWvT + (size_t)l * D_ * D_, D_, D_);
    TR(ca_w + ((size_t)l * 4 + 3) * D_ * D_, caWoT + (size_t)l * D_ * D_, D_, D_);
    TR(ffn_w1 + (size_t)l * D_ * F_, fW1T + (size_t)l * F_ * D_, D_, F_);
    TR(ffn_w2 + (size_t)l * F_ * D_, fW2T + (size_t)l * D_ * F_, F_, D_);
  }
  TR(out_w, outWT, D_, V_);
  TR(proj1_w, p1T, P_, 2 * D_);
  TR(proj2_w, p2T, 2 * D_, D_);
  k_f2b<<<dim3((B_ * P_) / 256), dim3(256), 0, stream>>>(protein, protb, B_ * P_);

  // ---- precompute: protein -> mem1; per-layer cross-attn constant ----
  GEMM(protb, p1T, proj1_b, nullptr, t1b, 2 * D_, P_, 1);     // relu(protein@proj1+b) [64,1024]
  GEMM(t1b, p2T, proj2_b, nullptr, mem1b, D_, 2 * D_, 0);     // mem1 [64,512]
  for (int l = 0; l < L_; ++l) {
    GEMM(mem1b, caWvT + (size_t)l * D_ * D_, ca_b + ((size_t)l * 4 + 2) * D_,
         nullptr, t1b, D_, D_, 0);
    GEMM(t1b, caWoT + (size_t)l * D_ * D_, ca_b + ((size_t)l * 4 + 3) * D_,
         caC + (size_t)l * B_ * D_, nullptr, D_, D_, 0);
  }

  // ---- greedy decode loop ----
  for (int t = 0; t < T_; ++t) {
    const int* tk = (t == 0) ? tok0 : curtok;
    k_embed<<<dim3(B_), dim3(D_), 0, stream>>>(emb, pe, tk, t, x, xb);
    for (int l = 0; l < L_; ++l) {
      // self-attn (single key): Wo(Wv x + bv) + bo
      GEMM(xb, saWvT + (size_t)l * D_ * D_, sa_b + ((size_t)l * 4 + 2) * D_,
           nullptr, t1b, D_, D_, 0);
      GEMM(t1b, saWoT + (size_t)l * D_ * D_, sa_b + ((size_t)l * 4 + 3) * D_,
           t2, nullptr, D_, D_, 0);
      k_add_ln<<<dim3(B_), dim3(256), 0, stream>>>(
          x, t2, ln_g + ((size_t)l * 3 + 0) * D_, ln_b + ((size_t)l * 3 + 0) * D_, x, xb);
      // cross-attn collapses to a per-(l,b) constant
      k_add_ln<<<dim3(B_), dim3(256), 0, stream>>>(
          x, caC + (size_t)l * B_ * D_,
          ln_g + ((size_t)l * 3 + 1) * D_, ln_b + ((size_t)l * 3 + 1) * D_, x, xb);
      // FFN
      GEMM(xb, fW1T + (size_t)l * F_ * D_, ffn_b1 + (size_t)l * F_,
           nullptr, t1b, F_, D_, 1);
      GEMM(t1b, fW2T + (size_t)l * D_ * F_, ffn_b2 + (size_t)l * D_,
           t2, nullptr, D_, F_, 0);
      k_add_ln<<<dim3(B_), dim3(256), 0, stream>>>(
          x, t2, ln_g + ((size_t)l * 3 + 2) * D_, ln_b + ((size_t)l * 3 + 2) * D_, x, xb);
    }
    GEMM(xb, outWT, out_b, logits, nullptr, V_, D_, 0);
    k_softmax_argmax<<<dim3(B_), dim3(256), 0, stream>>>(logits, probs, curtok, t);
  }
}
